// NemotronHMoEMLP_12360915878722
// MI455X (gfx1250) — matrix operand order repaired and checked by reading, not compiled, not run on hardware
//
#include <hip/hip_runtime.h>

// ---------------------------------------------------------------------------
// NemotronH MoE MLP for MI455X (gfx1250, wave32, WMMA + TDM async tiles).
// GEMMs: bf16 V_WMMA_F32_16X16X32_BF16, f32 accum. A/B tiles staged in LDS by
// the Tensor Data Mover (tensor_load_to_lds), double-buffered on TENSORcnt.
// ---------------------------------------------------------------------------

#define T_TOK 2048   // B*S tokens
#define HDIM  1024   // hidden
#define NEXP  8      // routed experts
#define IDIM  512    // moe_inter
#define SIDIM 2048   // shared_inter
#define NI    (NEXP * IDIM)   // 4096 flattened (e,i)

#define KCHUNK     128
#define ROW_PITCH  272                       // 256B row + 16B TDM pad (68 dwords)
#define TILE_BYTES (128 * ROW_PITCH)         // 34816 per tile buffer
#define SMEM_BYTES (4 * TILE_BYTES)          // A0,A1,B0,B1 = 139264

typedef __attribute__((ext_vector_type(16))) __bf16 v16bf;  // 8 VGPRs
typedef __attribute__((ext_vector_type(8)))  float  v8f;    // 8 VGPRs
typedef __attribute__((ext_vector_type(4)))  unsigned int v4u;
typedef __attribute__((ext_vector_type(8)))  int v8i;
typedef __attribute__((ext_vector_type(4)))  int v4i;

// ---- f32 -> bf16 round-to-nearest-even ------------------------------------
static __device__ __forceinline__ unsigned short f2bf(float f) {
  unsigned int u = __builtin_bit_cast(unsigned int, f);
  u += 0x7FFFu + ((u >> 16) & 1u);
  return (unsigned short)(u >> 16);
}

// ---------------------------------------------------------------------------
// TDM: issue a 2D tile load, 128 rows x 128 bf16, into LDS at lds_off.
// Row pitch in LDS becomes 272B via pad (interval=256B, amount=16B) so that
// ds_load_b128 fragment reads are bank-optimal.  D# per CDNA5 ISA ch.8.
// ---------------------------------------------------------------------------
static __device__ __forceinline__ void tdm_load_tile(unsigned lds_off,
                                                     const void* gptr,
                                                     unsigned row_stride_elems) {
  unsigned long long ga = (unsigned long long)gptr;
  v4u g0;
  g0[0] = 1u;                                              // count=1, user D#
  g0[1] = lds_off;                                         // lds_addr
  g0[2] = (unsigned)ga;                                    // global_addr[31:0]
  g0[3] = (unsigned)((ga >> 32) & 0x01FFFFFFu) | (2u << 30);  // [56:32] | type=2
  v8i g1;
  g1[0] = (int)((1u << 16)        // data_size = 1 -> 2 bytes
              | (1u << 20)        // pad_enable
              | (5u << 22)        // pad_interval: 64 dwords = 256B
              | (3u << 25));      // pad_amount: 4 dwords = 16B
  g1[1] = (int)(128u << 16);      // tensor_dim0[15:0] << 16   (dim0 = 128)
  g1[2] = (int)(128u << 16);      // dim0 hi=0 | tensor_dim1 lo (dim1 = 128)
  g1[3] = (int)(128u << 16);      // dim1 hi=0 | tile_dim0 = 128
  g1[4] = (int)128u;              // tile_dim1 = 128 | tile_dim2 = 0
  g1[5] = (int)row_stride_elems;  // tensor_dim0_stride[31:0]
  g1[6] = 0;                      // stride hi | tensor_dim1_stride lo
  g1[7] = 0;
  v4i gz4 = {0, 0, 0, 0};                     // group 2 (unused, 2D tensor)
  v8i gz8 = {0, 0, 0, 0, 0, 0, 0, 0};         // trailing group (unused)
  __builtin_amdgcn_tensor_load_to_lds(g0, g1, gz4, gz4, gz8, 0);
}

// ---- WMMA fragment loads from padded LDS tiles -----------------------------
// A (16x32): lane m   holds K{0..7},K{16..23}; lane m+16 K{8..15},K{24..31}
static __device__ __forceinline__ v16bf lds_frag_a(const char* buf, int row,
                                                   int ksub, int hi) {
  const char* p = buf + row * ROW_PITCH + (ksub + hi * 8) * 2;
  union { v16bf v; uint4 q[2]; } u;
  u.q[0] = *reinterpret_cast<const uint4*>(p);
  u.q[1] = *reinterpret_cast<const uint4*>(p + 32);
  return u.v;
}
// B (32x16): lane n holds column n K{0..7},K{16..23}; lane n+16 K{8..15},K{24..31}
static __device__ __forceinline__ v16bf lds_frag_b(const char* buf, int col,
                                                   int ksub, int hi) {
  const char* p = buf + col * ROW_PITCH + (ksub + hi * 8) * 2;
  union { v16bf v; uint4 q[2]; } u;
  u.q[0] = *reinterpret_cast<const uint4*>(p);
  u.q[1] = *reinterpret_cast<const uint4*>(p + 32);
  return u.v;
}

static __device__ __forceinline__ v8f wmma_bf16(v16bf a, v16bf b, v8f c) {
  return __builtin_amdgcn_wmma_f32_16x16x32_bf16(false, a, false, b, (short)0, c,
                                                 false, false);
}

// ---------------------------------------------------------------------------
// f32 -> bf16 bulk conversion (vectorized, grid-stride)
// ---------------------------------------------------------------------------
__global__ void __launch_bounds__(256)
cvt_bf16_kernel(const float* __restrict__ s, unsigned short* __restrict__ d, int n) {
  int i = (blockIdx.x * blockDim.x + threadIdx.x) * 4;
  int stride = gridDim.x * blockDim.x * 4;
  for (; i < n; i += stride) {
    float4 f = *reinterpret_cast<const float4*>(s + i);
    ushort4 o;
    o.x = f2bf(f.x); o.y = f2bf(f.y); o.z = f2bf(f.z); o.w = f2bf(f.w);
    *reinterpret_cast<ushort4*>(d + i) = o;
  }
}

// ---------------------------------------------------------------------------
// Router: sigmoid grouped top-k -> dense combine weights cw[T,8]
// ---------------------------------------------------------------------------
__global__ void __launch_bounds__(256)
router_kernel(const float* __restrict__ x,    // [T, H]
              const float* __restrict__ gw,   // [E, H]
              const float* __restrict__ gb,   // [E]
              float* __restrict__ cw) {       // [T, E]
  const int lane = threadIdx.x & 31;
  const int wid  = threadIdx.x >> 5;
  const int t    = blockIdx.x * 8 + wid;
  const float* xr = x + (size_t)t * HDIM;

  float acc[NEXP] = {0.f, 0.f, 0.f, 0.f, 0.f, 0.f, 0.f, 0.f};
  for (int k = lane; k < HDIM; k += 32) {
    float xv = xr[k];
#pragma unroll
    for (int e = 0; e < NEXP; ++e)
      acc[e] = fmaf(xv, gw[e * HDIM + k], acc[e]);
  }
#pragma unroll
  for (int e = 0; e < NEXP; ++e) {
#pragma unroll
    for (int off = 16; off > 0; off >>= 1)
      acc[e] += __shfl_xor(acc[e], off, 32);
  }

  if (lane == 0) {
    float scores[NEXP], sfc[NEXP];
#pragma unroll
    for (int e = 0; e < NEXP; ++e) {
      scores[e] = 1.0f / (1.0f + expf(-acc[e]));
      sfc[e]    = scores[e] + gb[e];
    }
    float gsc[2];
#pragma unroll
    for (int g = 0; g < 2; ++g) {
      float a = sfc[4 * g + 0], b = sfc[4 * g + 1];
      float c = sfc[4 * g + 2], d = sfc[4 * g + 3];
      float hi1 = fmaxf(a, b), lo1 = fminf(a, b);
      float hi2 = fmaxf(c, d), lo2 = fminf(c, d);
      gsc[g] = fmaxf(hi1, hi2) + fmaxf(fminf(hi1, hi2), fmaxf(lo1, lo2));
    }
    const int gsel = (gsc[1] > gsc[0]) ? 1 : 0;
    float masked[NEXP];
#pragma unroll
    for (int e = 0; e < NEXP; ++e)
      masked[e] = ((e >> 2) == gsel) ? sfc[e] : 0.0f;
    int i1 = 0; float b1 = -1e30f;
#pragma unroll
    for (int e = 0; e < NEXP; ++e) if (masked[e] > b1) { b1 = masked[e]; i1 = e; }
    int i2 = 0; float b2 = -1e30f;
#pragma unroll
    for (int e = 0; e < NEXP; ++e)
      if (e != i1 && masked[e] > b2) { b2 = masked[e]; i2 = e; }
    float w1 = scores[i1], w2 = scores[i2];
    float inv = 2.5f / (w1 + w2 + 1e-20f);
    w1 *= inv; w2 *= inv;
#pragma unroll
    for (int e = 0; e < NEXP; ++e)
      cw[(size_t)t * NEXP + e] = (e == i1) ? w1 : ((e == i2) ? w2 : 0.0f);
  }
}

// ---------------------------------------------------------------------------
// Up GEMM: D[M,N](bf16) = sqrelu( A[M,K] * B[N,K]^T ) [* cw[row, col>>9]]
// TDM double-buffered 128x128 tiles; 8 waves, each 32x64 (2x4 WMMA accums).
// ---------------------------------------------------------------------------
__global__ void __launch_bounds__(256)
gemm_up_kernel(const unsigned short* __restrict__ A,  // [M,K]
               const unsigned short* __restrict__ B,  // [N,K] row-major
               unsigned short* __restrict__ D,        // [M,N]
               const float* __restrict__ cw,          // [M,8] or nullptr
               int K, int N) {
  extern __shared__ char smem[];
  const int lane = threadIdx.x & 31;
  const int wid  = threadIdx.x >> 5;
  const int wm = wid & 3, wn = wid >> 2;
  const int l15 = lane & 15, hi = lane >> 4;
  const int mBlk = blockIdx.y * 128, nBlk = blockIdx.x * 128;
  const unsigned ldsOff = (unsigned)(unsigned long long)smem;

  const unsigned short* aG = A + (size_t)mBlk * K;
  const unsigned short* bG = B + (size_t)nBlk * K;
  const int nChunks = K / KCHUNK;

  if (wid == 0) {
    tdm_load_tile(ldsOff + 0 * TILE_BYTES, aG, (unsigned)K);
    tdm_load_tile(ldsOff + 2 * TILE_BYTES, bG, (unsigned)K);
    __builtin_amdgcn_s_wait_tensorcnt(0);
  }
  __syncthreads();

  v8f acc[2][4] = {};
  for (int c = 0; c < nChunks; ++c) {
    const int cur = c & 1;
    if (wid == 0 && c + 1 < nChunks) {
      const int nxt = (c + 1) & 1;
      const int k0 = (c + 1) * KCHUNK;
      tdm_load_tile(ldsOff + (0 + nxt) * TILE_BYTES, aG + k0, (unsigned)K);
      tdm_load_tile(ldsOff + (2 + nxt) * TILE_BYTES, bG + k0, (unsigned)K);
    }
    const char* aBuf = smem + (0 + cur) * TILE_BYTES;
    const char* bBuf = smem + (2 + cur) * TILE_BYTES;
#pragma unroll
    for (int ks = 0; ks < KCHUNK; ks += 32) {
      v16bf a0 = lds_frag_a(aBuf, wm * 32 + l15,      ks, hi);
      v16bf a1 = lds_frag_a(aBuf, wm * 32 + 16 + l15, ks, hi);
#pragma unroll
      for (int j = 0; j < 4; ++j) {
        v16bf b = lds_frag_b(bBuf, wn * 64 + j * 16 + l15, ks, hi);
        acc[0][j] = wmma_bf16(a0, b, acc[0][j]);
        acc[1][j] = wmma_bf16(a1, b, acc[1][j]);
      }
    }
    if (wid == 0) __builtin_amdgcn_s_wait_tensorcnt(0);
    __syncthreads();
  }

#pragma unroll
  for (int i = 0; i < 2; ++i) {
#pragma unroll
    for (int j = 0; j < 4; ++j) {
      const int col = nBlk + wn * 64 + j * 16 + l15;
      const int e = col >> 9;  // expert id (routed path)
#pragma unroll
      for (int r = 0; r < 8; ++r) {
        const int row = mBlk + wm * 32 + i * 16 + r + 8 * hi;
        float v = acc[i][j][r];
        v = fmaxf(v, 0.0f);
        v = v * v;
        if (cw) v *= cw[(size_t)row * NEXP + e];
        D[(size_t)row * N + col] = f2bf(v);
      }
    }
  }
}

// ---------------------------------------------------------------------------
// Down GEMM: Out[M,1024](f32) (=|+=) A[M,K](bf16) * W
// W element (k, col) at B + (k>>eShift)*eStride + col*ldB + (k & mask).
//   routed: eShift=9, eStride=H*I, ldB=I ; shared: eShift=30, ldB=SI
// KCHUNK=128 divides the 512-wide expert segments, so each TDM tile is
// a uniform 2D slab with row stride ldB.
// ---------------------------------------------------------------------------
__global__ void __launch_bounds__(256)
gemm_down_kernel(const unsigned short* __restrict__ A,  // [M,K]
                 const unsigned short* __restrict__ B,
                 float* __restrict__ Out,               // [M,1024]
                 int K, int ldB, int eShift, long long eStride,
                 int accumulate) {
  extern __shared__ char smem[];
  const int lane = threadIdx.x & 31;
  const int wid  = threadIdx.x >> 5;
  const int wm = wid & 3, wn = wid >> 2;
  const int l15 = lane & 15, hi = lane >> 4;
  const int mBlk = blockIdx.y * 128, nBlk = blockIdx.x * 128;
  const int eMask = (1 << eShift) - 1;
  const unsigned ldsOff = (unsigned)(unsigned long long)smem;

  const unsigned short* aG = A + (size_t)mBlk * K;
  const unsigned short* bG = B + (size_t)nBlk * ldB;
  const int nChunks = K / KCHUNK;

  if (wid == 0) {
    tdm_load_tile(ldsOff + 0 * TILE_BYTES, aG, (unsigned)K);
    tdm_load_tile(ldsOff + 2 * TILE_BYTES, bG, (unsigned)ldB);
    __builtin_amdgcn_s_wait_tensorcnt(0);
  }
  __syncthreads();

  v8f acc[2][4] = {};
  for (int c = 0; c < nChunks; ++c) {
    const int cur = c & 1;
    if (wid == 0 && c + 1 < nChunks) {
      const int nxt = (c + 1) & 1;
      const int k0 = (c + 1) * KCHUNK;
      const long long bo = (long long)(k0 >> eShift) * eStride + (k0 & eMask);
      tdm_load_tile(ldsOff + (0 + nxt) * TILE_BYTES, aG + k0, (unsigned)K);
      tdm_load_tile(ldsOff + (2 + nxt) * TILE_BYTES, bG + bo, (unsigned)ldB);
    }
    const char* aBuf = smem + (0 + cur) * TILE_BYTES;
    const char* bBuf = smem + (2 + cur) * TILE_BYTES;
#pragma unroll
    for (int ks = 0; ks < KCHUNK; ks += 32) {
      v16bf a0 = lds_frag_a(aBuf, wm * 32 + l15,      ks, hi);
      v16bf a1 = lds_frag_a(aBuf, wm * 32 + 16 + l15, ks, hi);
#pragma unroll
      for (int j = 0; j < 4; ++j) {
        v16bf b = lds_frag_b(bBuf, wn * 64 + j * 16 + l15, ks, hi);
        acc[0][j] = wmma_bf16(a0, b, acc[0][j]);
        acc[1][j] = wmma_bf16(a1, b, acc[1][j]);
      }
    }
    if (wid == 0) __builtin_amdgcn_s_wait_tensorcnt(0);
    __syncthreads();
  }

#pragma unroll
  for (int i = 0; i < 2; ++i) {
#pragma unroll
    for (int j = 0; j < 4; ++j) {
      const int col = nBlk + wn * 64 + j * 16 + l15;
#pragma unroll
      for (int r = 0; r < 8; ++r) {
        const int row = mBlk + wm * 32 + i * 16 + r + 8 * hi;
        const size_t o = (size_t)row * HDIM + col;
        float v = acc[i][j][r];
        if (accumulate) Out[o] += v; else Out[o] = v;
      }
    }
  }
}

// ---------------------------------------------------------------------------
// Host launcher
// ---------------------------------------------------------------------------
extern "C" void kernel_launch(void* const* d_in, const int* in_sizes, int n_in,
                              void* d_out, int out_size, void* d_ws, size_t ws_size,
                              hipStream_t stream) {
  const float* x   = (const float*)d_in[0];  // [B,S,H]
  const float* gw  = (const float*)d_in[1];  // [E,H]
  const float* gb  = (const float*)d_in[2];  // [E]
  const float* upw = (const float*)d_in[3];  // [E,I,H]
  const float* dnw = (const float*)d_in[4];  // [E,H,I]
  const float* suw = (const float*)d_in[5];  // [SI,H]
  const float* sdw = (const float*)d_in[6];  // [H,SI]
  float* out = (float*)d_out;                // [B,S,H] f32

  char* ws = (char*)d_ws;
  unsigned short* xb  = (unsigned short*)(ws + ((size_t)0  << 20));  //  4 MB
  unsigned short* upb = (unsigned short*)(ws + ((size_t)4  << 20));  //  8 MB
  unsigned short* dnb = (unsigned short*)(ws + ((size_t)12 << 20));  //  8 MB
  unsigned short* sub = (unsigned short*)(ws + ((size_t)20 << 20));  //  4 MB
  unsigned short* sdb = (unsigned short*)(ws + ((size_t)24 << 20));  //  4 MB
  unsigned short* r2w = (unsigned short*)(ws + ((size_t)28 << 20));  // 16 MB
  unsigned short* sh2 = (unsigned short*)(ws + ((size_t)44 << 20));  //  8 MB
  float*          cw  = (float*)         (ws + ((size_t)52 << 20));  // 64 KB

  auto cvt = [&](const float* s, unsigned short* d, int n) {
    cvt_bf16_kernel<<<dim3((n / 4 + 255) / 256), dim3(256), 0, stream>>>(s, d, n);
  };
  cvt(x,   xb,  T_TOK * HDIM);
  cvt(upw, upb, NEXP * IDIM * HDIM);
  cvt(dnw, dnb, NEXP * HDIM * IDIM);
  cvt(suw, sub, SIDIM * HDIM);
  cvt(sdw, sdb, HDIM * SIDIM);

  router_kernel<<<dim3(T_TOK / 8), dim3(256), 0, stream>>>(x, gw, gb, cw);

  // routed up: [T,1024] x [1024,4096] -> sqrelu*cw -> r2w (bf16)
  gemm_up_kernel<<<dim3(NI / 128, T_TOK / 128), dim3(256), SMEM_BYTES, stream>>>(
      xb, upb, r2w, cw, HDIM, NI);
  // shared up: [T,1024] x [1024,2048] -> sqrelu -> sh2 (bf16)
  gemm_up_kernel<<<dim3(SIDIM / 128, T_TOK / 128), dim3(256), SMEM_BYTES, stream>>>(
      xb, sub, sh2, nullptr, HDIM, SIDIM);
  // routed down: [T,4096] x flat(E,H,I) -> out (write)
  gemm_down_kernel<<<dim3(HDIM / 128, T_TOK / 128), dim3(256), SMEM_BYTES, stream>>>(
      r2w, dnb, out, NI, IDIM, 9, (long long)HDIM * IDIM, 0);
  // shared down: [T,2048] x [H,SI] -> out (+=)
  gemm_down_kernel<<<dim3(HDIM / 128, T_TOK / 128), dim3(256), SMEM_BYTES, stream>>>(
      sh2, sdb, out, SIDIM, SIDIM, 30, 0LL, 1);
}
